// MultiIPALayer_73332271612058
// MI455X (gfx1250) — matrix operand order repaired and checked by reading, not compiled, not run on hardware
//
#include <hip/hip_runtime.h>
#include <hip/hip_bf16.h>
#include <math.h>

#define BB 2
#define NN 512
#define CS 384
#define CZ 128
#define CH 16
#define HH 12
#define PQKN 4
#define PVN 8
#define CATW 2112
#define BN (BB*NN)

typedef __attribute__((ext_vector_type(16))) __bf16 bf16x16;
typedef __attribute__((ext_vector_type(8)))  float  f32x8;

// ---- fp32 -> bf16 (hardware packed cvt when available, else RNE software) ----
#if __has_builtin(__builtin_amdgcn_cvt_pk_bf16_f32)
__device__ inline unsigned pack2bf(float a, float b) {
  auto v = __builtin_amdgcn_cvt_pk_bf16_f32(a, b);   // low = a, high = b
  return __builtin_bit_cast(unsigned, v);
}
#else
__device__ inline unsigned pack2bf(float a, float b) {
  unsigned ua = __builtin_bit_cast(unsigned, a);
  unsigned ub = __builtin_bit_cast(unsigned, b);
  unsigned ra = (ua + 0x7FFFu + ((ua >> 16) & 1u)) >> 16;
  unsigned rb = (ub + 0x7FFFu + ((ub >> 16) & 1u)) & 0xFFFF0000u;
  return (ra & 0xFFFFu) | rb;
}
#endif
__device__ inline __bf16 f2bf(float x) {
  unsigned short h = (unsigned short)pack2bf(x, 0.f);
  return __builtin_bit_cast(__bf16, h);
}
__device__ inline f32x8 zero8() { f32x8 z; for (int i = 0; i < 8; ++i) z[i] = 0.f; return z; }
struct u4pair { uint4 a, b; };
__device__ inline bf16x16 frag_from(uint4 x, uint4 y) {
  u4pair p; p.a = x; p.b = y;
  return __builtin_bit_cast(bf16x16, p);
}
__device__ inline bf16x16 frag_zero() {
  uint4 z; z.x = z.y = z.z = z.w = 0u;
  return frag_from(z, z);
}
__device__ inline f32x8 wmma_bf(bf16x16 a, bf16x16 b, f32x8 c) {
  return __builtin_amdgcn_wmma_f32_16x16x32_bf16(false, a, false, b, (short)0, c, false, false);
}
// Fragment layouts used in this file (same contiguous k order for A and B), per lane:
//  A (16x32 MxK): row = lane&15; elem  e  -> K = e + 16*(lane>=16)  (one 32B run)
//  B (32x16 KxN): col = lane&15; elem  e  -> K = e + 16*(lane>=16)  (one 32B run)

// ---------------------------------------------------------------------------
// GEMM: C[M x Nc] = act(A * W + bias).  Requires M%64==0, K%32==0.
// 128 threads / 4 waves; 64x64 tile; 16 WMMAs per staging round.
// storeT: write C transposed as C[n * M + m].
// ---------------------------------------------------------------------------
__global__ void gemm_wmma(const float* __restrict__ A, const float* __restrict__ W,
                          const float* __restrict__ bias, float* __restrict__ C,
                          int M, int K, int Nc, int relu, int storeT)
{
  __shared__ __bf16 As[64][32];    // [m][k]
  __shared__ __bf16 Bst[64][32];   // transposed: [n][k]
  const int tid = threadIdx.x;
  const int wave = tid >> 5, lane = tid & 31;
  const int lm = lane & 15, lhi = (lane >> 4) & 1;
  const int m0 = blockIdx.x * 64;
  const int n0 = blockIdx.y * 64;
  const bool fullN = (n0 + 64 <= Nc);   // block-uniform
  f32x8 acc0 = zero8(), acc1 = zero8(), acc2 = zero8(), acc3 = zero8();

  for (int k0 = 0; k0 < K; k0 += 32) {
    #pragma unroll
    for (int t = 0; t < 4; ++t) {
      int f = tid + 128 * t;              // 0..511 float4 slots of 64x32 tile
      int r = f >> 3, kk = (f & 7) * 4;
      const float4 v = *(const float4*)(A + (size_t)(m0 + r) * K + (k0 + kk));
      unsigned* dst = (unsigned*)&As[r][kk];
      dst[0] = pack2bf(v.x, v.y);
      dst[1] = pack2bf(v.z, v.w);
    }
    if (fullN) {
      #pragma unroll
      for (int t = 0; t < 4; ++t) {
        int f = tid + 128 * t;            // 0..511 float4 slots of 32x64 tile
        int r = f >> 4, c = (f & 15) * 4;
        const float4 v = *(const float4*)(W + (size_t)(k0 + r) * Nc + (n0 + c));
        Bst[c + 0][r] = f2bf(v.x);
        Bst[c + 1][r] = f2bf(v.y);
        Bst[c + 2][r] = f2bf(v.z);
        Bst[c + 3][r] = f2bf(v.w);
      }
    } else {
      #pragma unroll
      for (int t = 0; t < 16; ++t) {
        int idx = tid + 128 * t;          // 0..2047
        int r = idx >> 6, c = idx & 63;
        int gn = n0 + c;
        int gc = gn < Nc ? gn : (Nc - 1); // clamped load, no exec-mask branch
        float v = W[(size_t)(k0 + r) * Nc + gc];
        if (gn >= Nc) v = 0.f;
        Bst[c][r] = f2bf(v);
      }
    }
    if (k0 + 32 < K)
      __builtin_prefetch(A + (size_t)(m0 + (tid >> 1)) * K + (k0 + 32), 0, 1);
    __syncthreads();

    const uint4* ap = (const uint4*)&As[wave * 16 + lm][lhi * 16];
    bf16x16 af = frag_from(ap[0], ap[1]);
    {
      const uint4* bp = (const uint4*)&Bst[0 * 16 + lm][lhi * 16];
      acc0 = wmma_bf(af, frag_from(bp[0], bp[1]), acc0);
    }
    {
      const uint4* bp = (const uint4*)&Bst[1 * 16 + lm][lhi * 16];
      acc1 = wmma_bf(af, frag_from(bp[0], bp[1]), acc1);
    }
    {
      const uint4* bp = (const uint4*)&Bst[2 * 16 + lm][lhi * 16];
      acc2 = wmma_bf(af, frag_from(bp[0], bp[1]), acc2);
    }
    {
      const uint4* bp = (const uint4*)&Bst[3 * 16 + lm][lhi * 16];
      acc3 = wmma_bf(af, frag_from(bp[0], bp[1]), acc3);
    }
    __syncthreads();
  }

  f32x8 accs[4] = {acc0, acc1, acc2, acc3};
  #pragma unroll
  for (int nt = 0; nt < 4; ++nt) {
    #pragma unroll
    for (int r = 0; r < 8; ++r) {
      int gm = m0 + wave * 16 + r + (lhi << 3);
      int gn = n0 + nt * 16 + lm;
      if (gn < Nc) {
        float v = accs[nt][r] + bias[gn];
        if (relu) v = v > 0.f ? v : 0.f;
        if (storeT) C[(size_t)gn * M + gm] = v;
        else        C[(size_t)gm * Nc + gn] = v;
      }
    }
  }
}

// ---------------------------------------------------------------------------
__global__ void init_kernel(const float* s, const float* rot, const float* trans,
                            float* s_ws, float* rot_ws, float* trans_ws)
{
  int i = blockIdx.x * blockDim.x + threadIdx.x;
  if (i < BN * CS) s_ws[i] = s[i];
  if (i < BN * 9)  rot_ws[i] = rot[i];
  if (i < BN * 3)  trans_ws[i] = trans[i];
}

// qp/kp/vp -> global frame: g = R p + t
__global__ void to_global_kernel(const float* __restrict__ qpbuf, const float* __restrict__ kvpbuf,
                                 const float* __restrict__ rot, const float* __restrict__ trans,
                                 float* qp_g, float* kp_g, float* vp_g)
{
  int idx = blockIdx.x * blockDim.x + threadIdx.x;
  if (idx >= BN * HH * 16) return;
  int pp = idx & 15;
  int h  = (idx >> 4) % HH;
  int bn = idx / (16 * HH);
  const float* R = rot + (size_t)bn * 9;
  const float* t = trans + (size_t)bn * 3;
  float p0, p1, p2;
  if (pp < PQKN) {
    const float* src = qpbuf + (size_t)bn * (HH * PQKN * 3) + (h * PQKN + pp) * 3;
    p0 = src[0]; p1 = src[1]; p2 = src[2];
  } else {
    const float* src = kvpbuf + (size_t)bn * (HH * (PQKN + PVN) * 3) + (h * (PQKN + PVN) + (pp - PQKN)) * 3;
    p0 = src[0]; p1 = src[1]; p2 = src[2];
  }
  float g0 = R[0]*p0 + R[1]*p1 + R[2]*p2 + t[0];
  float g1 = R[3]*p0 + R[4]*p1 + R[5]*p2 + t[1];
  float g2 = R[6]*p0 + R[7]*p1 + R[8]*p2 + t[2];
  if (pp < PQKN) {
    float* d = qp_g + (((size_t)bn * HH + h) * PQKN + pp) * 3;
    d[0] = g0; d[1] = g1; d[2] = g2;
  } else if (pp < 2 * PQKN) {
    float* d = kp_g + (((size_t)bn * HH + h) * PQKN + (pp - PQKN)) * 3;
    d[0] = g0; d[1] = g1; d[2] = g2;
  } else {
    float* d = vp_g + (((size_t)bn * HH + h) * PVN + (pp - 2 * PQKN)) * 3;
    d[0] = g0; d[1] = g1; d[2] = g2;
  }
}

// qq[b,h,n] = sum_p |qp|^2 ; kk likewise
__global__ void ptsq_kernel(const float* __restrict__ qp_g, const float* __restrict__ kp_g,
                            float* qq, float* kk)
{
  int idx = blockIdx.x * blockDim.x + threadIdx.x;
  if (idx >= BB * HH * NN) return;
  int n = idx % NN;
  int h = (idx / NN) % HH;
  int b = idx / (NN * HH);
  int bn = b * NN + n;
  const float* p = qp_g + ((size_t)bn * HH + h) * (PQKN * 3);
  float s = 0.f;
  for (int d = 0; d < PQKN * 3; ++d) s += p[d] * p[d];
  qq[idx] = s;
  p = kp_g + ((size_t)bn * HH + h) * (PQKN * 3);
  s = 0.f;
  for (int d = 0; d < PQKN * 3; ++d) s += p[d] * p[d];
  kk[idx] = s;
}

// ---------------------------------------------------------------------------
// Per-(b,h,n):  qhat/khat rows (augmented K=32 for score WMMA) and
// vhat[b][h][feat(48)][j] transposed value rows for the apply WMMA:
//   feats 0:16 = v, 16:40 = vp_g (p,x flattened), 40:48 = zero.
// ---------------------------------------------------------------------------
__global__ void build_hat_kernel(const float* __restrict__ q, const float* __restrict__ kvbuf,
                                 const float* __restrict__ qp_g, const float* __restrict__ kp_g,
                                 const float* __restrict__ vp_g, const float* __restrict__ hw_l,
                                 __bf16* __restrict__ qhat, __bf16* __restrict__ khat,
                                 __bf16* __restrict__ vhat)
{
  int idx = blockIdx.x * blockDim.x + threadIdx.x;   // (b*HH + h)*NN + n
  if (idx >= BB * HH * NN) return;
  int n = idx % NN;
  int h = (idx / NN) % HH;
  int b = idx / (NN * HH);
  int bn = b * NN + n;
  float headw = log1pf(__expf(hw_l[h])) * 0.13608276f;   // softplus(hw) * sqrt(1/54)
  const float* qs  = q     + ((size_t)bn * HH + h) * CH;
  const float* ks  = kvbuf + (((size_t)bn * HH + h) * 2 + 0) * CH;
  const float* vs  = kvbuf + (((size_t)bn * HH + h) * 2 + 1) * CH;
  const float* qps = qp_g + ((size_t)bn * HH + h) * (PQKN * 3);
  const float* kps = kp_g + ((size_t)bn * HH + h) * (PQKN * 3);
  const float* vps = vp_g + ((size_t)bn * HH + h) * (PVN * 3);
  float qv[32], kv[32];
  #pragma unroll
  for (int c = 0; c < CH; ++c) { qv[c] = qs[c] * 0.14433757f; kv[c] = ks[c]; }
  #pragma unroll
  for (int d = 0; d < PQKN * 3; ++d) { qv[CH + d] = qps[d] * headw; kv[CH + d] = kps[d]; }
  #pragma unroll
  for (int d = 28; d < 32; ++d) { qv[d] = 0.f; kv[d] = 0.f; }
  unsigned* qd = (unsigned*)(qhat + (size_t)idx * 32);
  unsigned* kd = (unsigned*)(khat + (size_t)idx * 32);
  #pragma unroll
  for (int c = 0; c < 16; ++c) {
    qd[c] = pack2bf(qv[2 * c], qv[2 * c + 1]);
    kd[c] = pack2bf(kv[2 * c], kv[2 * c + 1]);
  }
  // transposed value rows (column n, feature-major)
  __bf16* vd = vhat + ((size_t)(b * HH + h) * 48) * NN + n;
  #pragma unroll
  for (int c = 0; c < CH; ++c)       vd[(size_t)c * NN]        = f2bf(vs[c]);
  #pragma unroll
  for (int d = 0; d < PVN * 3; ++d)  vd[(size_t)(CH + d) * NN] = f2bf(vps[d]);
  #pragma unroll
  for (int d = 40; d < 48; ++d)      vd[(size_t)d * NN]        = f2bf(0.f);
}

// ---------------------------------------------------------------------------
// Attention scores (WMMA on qhat/khat) + softmax.  grid (NN/16, HH, BB), 128 thr.
// biasT layout [h][b][i][j] -> contiguous j reads.
// ---------------------------------------------------------------------------
__global__ void attn_kernel(const __bf16* __restrict__ qhat, const __bf16* __restrict__ khat,
                            const float* __restrict__ qq, const float* __restrict__ kk,
                            const float* __restrict__ biasT, const float* __restrict__ mask,
                            const float* __restrict__ hw_l, __bf16* __restrict__ a_out)
{
  const int b = blockIdx.z, h = blockIdx.y, it = blockIdx.x;
  const int tid = threadIdx.x, wave = tid >> 5, lane = tid & 31;
  const int lm = lane & 15, lhi = (lane >> 4) & 1;
  __shared__ float sc[16][NN];
  __shared__ float red[16][8];
  __shared__ float rowmax[16];
  __shared__ float rowsum[16];

  const float headw = log1pf(__expf(hw_l[h])) * 0.13608276f;
  const size_t bh = (size_t)b * HH + h;

  const uint4* qr = (const uint4*)(qhat + (bh * NN + it * 16 + lm) * 32 + lhi * 16);
  bf16x16 af = frag_from(qr[0], qr[1]);

  for (int jt = wave; jt < NN / 16; jt += 4) {
    const uint4* kr = (const uint4*)(khat + (bh * NN + jt * 16 + lm) * 32 + lhi * 16);
    bf16x16 bfr = frag_from(kr[0], kr[1]);
    f32x8 acc = wmma_bf(af, bfr, zero8());
    #pragma unroll
    for (int r = 0; r < 8; ++r) {
      int iloc = r + (lhi << 3);
      int i = it * 16 + iloc;
      int j = jt * 16 + lm;
      float bterm = biasT[(bh * NN + i) * NN + j];   // b*(H*N*N) + h*N*N + i*N + j
      float v = acc[r] + 0.57735027f * bterm
              - 0.5f * headw * (qq[bh * NN + i] + kk[bh * NN + j])
              + (mask[b * NN + i] * mask[b * NN + j] - 1.f) * 1e9f;
      sc[iloc][j] = v;
    }
  }
  __syncthreads();

  int row = tid >> 3, sub = tid & 7;
  float pm = -1e30f;
  for (int j = sub; j < NN; j += 8) pm = fmaxf(pm, sc[row][j]);
  red[row][sub] = pm;
  __syncthreads();
  if (sub == 0) {
    float m = red[row][0];
    for (int t = 1; t < 8; ++t) m = fmaxf(m, red[row][t]);
    rowmax[row] = m;
  }
  __syncthreads();
  float rm = rowmax[row];
  float ps = 0.f;
  for (int j = sub; j < NN; j += 8) {
    float e = __expf(sc[row][j] - rm);
    sc[row][j] = e;
    ps += e;
  }
  red[row][sub] = ps;
  __syncthreads();
  if (sub == 0) {
    float s = 0.f;
    for (int t = 0; t < 8; ++t) s += red[row][t];
    rowsum[row] = s;
  }
  __syncthreads();
  for (int idx = tid; idx < 16 * (NN / 2); idx += 128) {
    int iloc = idx / (NN / 2);
    int j = (idx % (NN / 2)) * 2;
    float rs = 1.f / rowsum[iloc];
    unsigned pk = pack2bf(sc[iloc][j] * rs, sc[iloc][j + 1] * rs);
    *(unsigned*)(a_out + (bh * NN + it * 16 + iloc) * NN + j) = pk;
  }
}

// ---------------------------------------------------------------------------
// o = a@v and op = a@vp via transposed vhat rows: all-b128 fragments.
// grid (NN/64, HH, BB), 128 thr; wave w handles i-tile blockIdx.x*4+w.
// ---------------------------------------------------------------------------
__global__ void attn_ov_kernel(const __bf16* __restrict__ a_bf, const __bf16* __restrict__ vhat,
                               float* __restrict__ cat, float* __restrict__ op_raw)
{
  const int b = blockIdx.z, h = blockIdx.y;
  const int tid = threadIdx.x, wave = tid >> 5, lane = tid & 31;
  const int it = blockIdx.x * 4 + wave;
  const int lm = lane & 15, lhi = (lane >> 4) & 1;
  const size_t bh = (size_t)b * HH + h;
  f32x8 acc0 = zero8(), acc1 = zero8(), acc2 = zero8();
  const __bf16* vrow0 = vhat + (bh * 48 + 0 * 16 + lm) * NN;
  const __bf16* vrow1 = vhat + (bh * 48 + 1 * 16 + lm) * NN;
  const __bf16* vrow2 = vhat + (bh * 48 + 2 * 16 + lm) * NN;

  for (int kt = 0; kt < NN / 32; ++kt) {
    const uint4* ar = (const uint4*)(a_bf + (bh * NN + it * 16 + lm) * NN + kt * 32 + lhi * 16);
    bf16x16 af = frag_from(ar[0], ar[1]);
    int j0 = kt * 32 + lhi * 16;
    const uint4* b0 = (const uint4*)(vrow0 + j0);
    const uint4* b1 = (const uint4*)(vrow1 + j0);
    const uint4* b2 = (const uint4*)(vrow2 + j0);
    acc0 = wmma_bf(af, frag_from(b0[0], b0[1]), acc0);
    acc1 = wmma_bf(af, frag_from(b1[0], b1[1]), acc1);
    acc2 = wmma_bf(af, frag_from(b2[0], b2[1]), acc2);
  }
  #pragma unroll
  for (int r = 0; r < 8; ++r) {
    int i = it * 16 + r + (lhi << 3);
    size_t bn = (size_t)b * NN + i;
    cat[bn * CATW + h * CH + lm] = acc0[r];
    float* op = op_raw + (bn * HH + h) * (PVN * 3);
    op[lm] = acc1[r];                    // d = 0..15
    if (lm < 8) op[16 + lm] = acc2[r];   // d = 16..23
  }
}

// ---------------------------------------------------------------------------
// o_pair = a @ z per (b,i): [HH pad16 x 512] @ [512 x 128].  grid (NN, BB), 4 waves.
// Streams full fp32 z once per layer (lane-coalesced scalar loads == the bytes).
// ---------------------------------------------------------------------------
__global__ void attn_pair_kernel(const __bf16* __restrict__ a_bf, const float* __restrict__ z,
                                 float* __restrict__ cat)
{
  const int i = blockIdx.x, b = blockIdx.y;
  const int tid = threadIdx.x, wave = tid >> 5, lane = tid & 31;
  const int lm = lane & 15, lhi = (lane >> 4) & 1;
  const int ntA = wave * 2, ntB = wave * 2 + 1;
  f32x8 accA = zero8(), accB = zero8();
  const int hcl = lm < HH ? lm : (HH - 1);

  for (int kt = 0; kt < NN / 32; ++kt) {
    const uint4* ar = (const uint4*)(a_bf + (((size_t)b * HH + hcl) * NN + i) * NN + kt * 32 + lhi * 16);
    bf16x16 af = frag_from(ar[0], ar[1]);
    if (lm >= HH) af = frag_zero();
    bf16x16 bA, bB;
    #pragma unroll
    for (int e = 0; e < 16; ++e) {
      int j = kt * 32 + e + (lhi << 4);
      size_t zrow = ((size_t)(b * NN + i) * NN + j) * CZ;
      bA[e] = f2bf(z[zrow + ntA * 16 + lm]);
      bB[e] = f2bf(z[zrow + ntB * 16 + lm]);
    }
    accA = wmma_bf(af, bA, accA);
    accB = wmma_bf(af, bB, accB);
  }
  size_t base = ((size_t)b * NN + i) * CATW + (CH * HH + 4 * PVN * HH);   // 576
  #pragma unroll
  for (int r = 0; r < 8; ++r) {
    int h = r + (lhi << 3);
    if (h < HH) {
      cat[base + (size_t)h * CZ + ntA * 16 + lm] = accA[r];
      cat[base + (size_t)h * CZ + ntB * 16 + lm] = accB[r];
    }
  }
}

// op -> local frame: R^T (op - t); write xyz + norm into cat
__global__ void op_finalize_kernel(const float* __restrict__ op_raw, const float* __restrict__ rot,
                                   const float* __restrict__ trans, float* __restrict__ cat)
{
  int idx = blockIdx.x * blockDim.x + threadIdx.x;
  if (idx >= BN * HH * PVN) return;
  int p = idx % PVN;
  int h = (idx / PVN) % HH;
  int bn = idx / (PVN * HH);
  const float* R = rot + (size_t)bn * 9;
  const float* t = trans + (size_t)bn * 3;
  const float* o = op_raw + (((size_t)bn * HH + h) * PVN + p) * 3;
  float v0 = o[0] - t[0], v1 = o[1] - t[1], v2 = o[2] - t[2];
  float l0 = R[0]*v0 + R[3]*v1 + R[6]*v2;
  float l1 = R[1]*v0 + R[4]*v1 + R[7]*v2;
  float l2 = R[2]*v0 + R[5]*v1 + R[8]*v2;
  float nrm = sqrtf(l0*l0 + l1*l1 + l2*l2 + 1e-8f);
  int hp = h * PVN + p;
  float* c = cat + (size_t)bn * CATW;
  c[CH * HH + 0 * (HH * PVN) + hp] = l0;
  c[CH * HH + 1 * (HH * PVN) + hp] = l1;
  c[CH * HH + 2 * (HH * PVN) + hp] = l2;
  c[CH * HH + 3 * (HH * PVN) + hp] = nrm;
}

// s = LayerNorm(s + add * (use_mask ? mask : 1)); one block per row
__global__ void res_ln_kernel(float* __restrict__ s, const float* __restrict__ add,
                              const float* __restrict__ mask, const float* __restrict__ g,
                              const float* __restrict__ bb, int use_mask)
{
  int bn = blockIdx.x;
  int tid = threadIdx.x;
  __shared__ float red[128];
  __shared__ float buf[CS];
  float* srow = s + (size_t)bn * CS;
  const float* arow = add + (size_t)bn * CS;
  float mk = use_mask ? mask[bn] : 1.f;
  float ls = 0.f;
  for (int c = tid; c < CS; c += 128) { float v = srow[c] + arow[c] * mk; buf[c] = v; ls += v; }
  red[tid] = ls; __syncthreads();
  for (int st = 64; st > 0; st >>= 1) { if (tid < st) red[tid] += red[tid + st]; __syncthreads(); }
  float mu = red[0] / (float)CS;
  __syncthreads();
  float lv = 0.f;
  for (int c = tid; c < CS; c += 128) { float d = buf[c] - mu; lv += d * d; }
  red[tid] = lv; __syncthreads();
  for (int st = 64; st > 0; st >>= 1) { if (tid < st) red[tid] += red[tid + st]; __syncthreads(); }
  float inv = rsqrtf(red[0] / (float)CS + 1e-5f);
  for (int c = tid; c < CS; c += 128) srow[c] = (buf[c] - mu) * inv * g[c] + bb[c];
}

// quaternion backbone update
__global__ void backbone_kernel(float* __restrict__ rot, float* __restrict__ trans,
                                const float* __restrict__ upd)
{
  int bn = blockIdx.x * blockDim.x + threadIdx.x;
  if (bn >= BN) return;
  const float* u = upd + (size_t)bn * 6;
  float inv = rsqrtf(1.f + u[0]*u[0] + u[1]*u[1] + u[2]*u[2]);
  float w = inv, x = u[0]*inv, y = u[1]*inv, zq = u[2]*inv;
  float Rn[9] = {1.f-2.f*(y*y+zq*zq), 2.f*(x*y - w*zq),     2.f*(x*zq + w*y),
                 2.f*(x*y + w*zq),    1.f-2.f*(x*x+zq*zq),  2.f*(y*zq - w*x),
                 2.f*(x*zq - w*y),    2.f*(y*zq + w*x),     1.f-2.f*(x*x+y*y)};
  float* R = rot + (size_t)bn * 9;
  float* t = trans + (size_t)bn * 3;
  t[0] += R[0]*u[3] + R[1]*u[4] + R[2]*u[5];
  t[1] += R[3]*u[3] + R[4]*u[4] + R[5]*u[5];
  t[2] += R[6]*u[3] + R[7]*u[4] + R[8]*u[5];
  float out[9];
  for (int i = 0; i < 3; ++i)
    for (int j = 0; j < 3; ++j)
      out[i*3+j] = R[i*3+0]*Rn[0*3+j] + R[i*3+1]*Rn[1*3+j] + R[i*3+2]*Rn[2*3+j];
  for (int k = 0; k < 9; ++k) R[k] = out[k];
}

__global__ void final_kernel(const float* __restrict__ rot, const float* __restrict__ trans,
                             float* __restrict__ out)
{
  int bn = blockIdx.x * blockDim.x + threadIdx.x;
  if (bn >= BN) return;
  for (int k = 0; k < 9; ++k) out[(size_t)bn * 12 + k] = rot[(size_t)bn * 9 + k];
  for (int k = 0; k < 3; ++k) out[(size_t)bn * 12 + 9 + k] = trans[(size_t)bn * 3 + k];
}

// ---------------------------------------------------------------------------
extern "C" void kernel_launch(void* const* d_in, const int* in_sizes, int n_in,
                              void* d_out, int out_size, void* d_ws, size_t ws_size,
                              hipStream_t stream)
{
  (void)in_sizes; (void)n_in; (void)out_size; (void)ws_size;
  const float* rot_i   = (const float*)d_in[0];
  const float* trans_i = (const float*)d_in[1];
  const float* s_i     = (const float*)d_in[2];
  const float* z_i     = (const float*)d_in[3];
  const float* mask_i  = (const float*)d_in[4];
  const float* Wq   = (const float*)d_in[5];  const float* bq   = (const float*)d_in[6];
  const float* Wkv  = (const float*)d_in[7];  const float* bkv  = (const float*)d_in[8];
  const float* Wqp  = (const float*)d_in[9];  const float* bqp  = (const float*)d_in[10];
  const float* Wkvp = (const float*)d_in[11]; const float* bkvp = (const float*)d_in[12];
  const float* Wb   = (const float*)d_in[13]; const float* bbp  = (const float*)d_in[14];
  const float* hw   = (const float*)d_in[15];
  const float* Wo   = (const float*)d_in[16]; const float* bo   = (const float*)d_in[17];
  const float* lng  = (const float*)d_in[18]; const float* lnb  = (const float*)d_in[19];
  const float* Wt1  = (const float*)d_in[20]; const float* bt1  = (const float*)d_in[21];
  const float* Wt2  = (const float*)d_in[22]; const float* bt2  = (const float*)d_in[23];
  const float* Wt3  = (const float*)d_in[24]; const float* bt3  = (const float*)d_in[25];
  const float* tlng = (const float*)d_in[26]; const float* tlnb = (const float*)d_in[27];
  const float* Wbb  = (const float*)d_in[28]; const float* bbb  = (const float*)d_in[29];

  char* wsb = (char*)d_ws;
  size_t off = 0;
  auto alloc = [&](size_t bytes) -> void* {
    void* p = (void*)(wsb + off);
    off += (bytes + 255) & ~(size_t)255;
    return p;
  };
  float* s_ws    = (float*)alloc((size_t)BN * CS * 4);
  float* rot_ws  = (float*)alloc((size_t)BN * 9 * 4);
  float* trans_ws= (float*)alloc((size_t)BN * 3 * 4);
  float* qbuf    = (float*)alloc((size_t)BN * HH * CH * 4);
  float* kvbuf   = (float*)alloc((size_t)BN * HH * 2 * CH * 4);
  float* qpbuf   = (float*)alloc((size_t)BN * HH * PQKN * 3 * 4);
  float* kvpbuf  = (float*)alloc((size_t)BN * HH * (PQKN + PVN) * 3 * 4);
  float* qp_g    = (float*)alloc((size_t)BN * HH * PQKN * 3 * 4);
  float* kp_g    = (float*)alloc((size_t)BN * HH * PQKN * 3 * 4);
  float* vp_g    = (float*)alloc((size_t)BN * HH * PVN * 3 * 4);
  float* qqb     = (float*)alloc((size_t)BB * HH * NN * 4);
  float* kkb     = (float*)alloc((size_t)BB * HH * NN * 4);
  float* biasT   = (float*)alloc((size_t)HH * BB * NN * NN * 4);
  __bf16* qhat   = (__bf16*)alloc((size_t)BB * HH * NN * 32 * 2);
  __bf16* khat   = (__bf16*)alloc((size_t)BB * HH * NN * 32 * 2);
  __bf16* vhat   = (__bf16*)alloc((size_t)BB * HH * 48 * NN * 2);
  __bf16* a_bf   = (__bf16*)alloc((size_t)BB * HH * NN * NN * 2);
  float* catb    = (float*)alloc((size_t)BN * CATW * 4);
  float* op_raw  = (float*)alloc((size_t)BN * HH * PVN * 3 * 4);
  float* emb     = (float*)alloc((size_t)BN * CS * 4);
  float* t1b     = (float*)alloc((size_t)BN * CS * 4);
  float* t2b     = (float*)alloc((size_t)BN * CS * 4);
  float* t3b     = (float*)alloc((size_t)BN * CS * 4);
  float* updb    = (float*)alloc((size_t)BN * 6 * 4);

  init_kernel<<<(BN * CS + 255) / 256, 256, 0, stream>>>(s_i, rot_i, trans_i, s_ws, rot_ws, trans_ws);

  for (int l = 0; l < 4; ++l) {
    // projections from s  (M=1024, K=384)
    gemm_wmma<<<dim3(BN / 64, (HH * CH + 63) / 64), 128, 0, stream>>>(
        s_ws, Wq + (size_t)l * CS * HH * CH, bq + (size_t)l * HH * CH, qbuf, BN, CS, HH * CH, 0, 0);
    gemm_wmma<<<dim3(BN / 64, (HH * 2 * CH + 63) / 64), 128, 0, stream>>>(
        s_ws, Wkv + (size_t)l * CS * HH * 2 * CH, bkv + (size_t)l * HH * 2 * CH, kvbuf, BN, CS, HH * 2 * CH, 0, 0);
    gemm_wmma<<<dim3(BN / 64, (HH * PQKN * 3 + 63) / 64), 128, 0, stream>>>(
        s_ws, Wqp + (size_t)l * CS * HH * PQKN * 3, bqp + (size_t)l * HH * PQKN * 3, qpbuf, BN, CS, HH * PQKN * 3, 0, 0);
    gemm_wmma<<<dim3(BN / 64, (HH * (PQKN + PVN) * 3 + 63) / 64), 128, 0, stream>>>(
        s_ws, Wkvp + (size_t)l * CS * HH * (PQKN + PVN) * 3, bkvp + (size_t)l * HH * (PQKN + PVN) * 3,
        kvpbuf, BN, CS, HH * (PQKN + PVN) * 3, 0, 0);
    // frame transforms, |pt|^2, augmented q/k/v rows
    to_global_kernel<<<(BN * HH * 16 + 255) / 256, 256, 0, stream>>>(
        qpbuf, kvpbuf, rot_ws, trans_ws, qp_g, kp_g, vp_g);
    ptsq_kernel<<<(BB * HH * NN + 255) / 256, 256, 0, stream>>>(qp_g, kp_g, qqb, kkb);
    build_hat_kernel<<<(BB * HH * NN + 255) / 256, 256, 0, stream>>>(
        qbuf, kvbuf, qp_g, kp_g, vp_g, hw + (size_t)l * HH, qhat, khat, vhat);
    // pair bias: z @ Wb, stored transposed -> [h][b][i][j]   (streams z once)
    gemm_wmma<<<dim3((BB * NN * NN) / 64, 1), 128, 0, stream>>>(
        z_i, Wb + (size_t)l * CZ * HH, bbp + (size_t)l * HH, biasT, BB * NN * NN, CZ, HH, 0, 1);
    // attention scores + softmax
    attn_kernel<<<dim3(NN / 16, HH, BB), 128, 0, stream>>>(
        qhat, khat, qqb, kkb, biasT, mask_i, hw + (size_t)l * HH, a_bf);
    // attention applies
    attn_ov_kernel<<<dim3(NN / 64, HH, BB), 128, 0, stream>>>(a_bf, vhat, catb, op_raw);
    attn_pair_kernel<<<dim3(NN, BB), 128, 0, stream>>>(a_bf, z_i, catb);   // streams z once
    op_finalize_kernel<<<(BN * HH * PVN + 255) / 256, 256, 0, stream>>>(op_raw, rot_ws, trans_ws, catb);
    // output projection  (K=2112)
    gemm_wmma<<<dim3(BN / 64, (CS + 63) / 64), 128, 0, stream>>>(
        catb, Wo + (size_t)l * CATW * CS, bo + (size_t)l * CS, emb, BN, CATW, CS, 0, 0);
    res_ln_kernel<<<BN, 128, 0, stream>>>(s_ws, emb, mask_i, lng, lnb, 1);
    // transition
    gemm_wmma<<<dim3(BN / 64, (CS + 63) / 64), 128, 0, stream>>>(s_ws, Wt1, bt1, t1b, BN, CS, CS, 1, 0);
    gemm_wmma<<<dim3(BN / 64, (CS + 63) / 64), 128, 0, stream>>>(t1b, Wt2, bt2, t2b, BN, CS, CS, 1, 0);
    gemm_wmma<<<dim3(BN / 64, (CS + 63) / 64), 128, 0, stream>>>(t2b, Wt3, bt3, t3b, BN, CS, CS, 0, 0);
    res_ln_kernel<<<BN, 128, 0, stream>>>(s_ws, t3b, mask_i, tlng, tlnb, 0);
    // backbone update
    gemm_wmma<<<dim3(BN / 64, 1), 128, 0, stream>>>(s_ws, Wbb, bbb, updb, BN, CS, 6, 0, 0);
    backbone_kernel<<<(BN + 255) / 256, 256, 0, stream>>>(rot_ws, trans_ws, updb);
  }
  final_kernel<<<(BN + 255) / 256, 256, 0, stream>>>(rot_ws, trans_ws, (float*)d_out);
}
